// NonLocal2D_80788334838025
// MI455X (gfx1250) — matrix operand order repaired and checked by reading, not compiled, not run on hardware
//
#include <hip/hip_runtime.h>
#include <hip/hip_bf16.h>
#include <math.h>

#define C_IN   256
#define CI_N   128
#define NB     4
#define S_TOT  4096
#define ATT_SCALE 11.313708498984761f   // sqrt(128): reference divides by d**-0.5

typedef __attribute__((ext_vector_type(16))) __bf16 v16bf;
typedef __attribute__((ext_vector_type(8)))  float  v8f;

union Frag16 { v16bf v; __bf16 e[16]; uint4 q[2]; };

__device__ __forceinline__ v8f wmma_bf16(v16bf a, v16bf b, v8f c) {
  // (neg_a, A, neg_b, B, c_mod, C, reuse_a, reuse_b)
  return __builtin_amdgcn_wmma_f32_16x16x32_bf16(false, a, false, b, (short)0, c,
                                                 false, false);
}

__device__ __forceinline__ v8f zero8() {
  v8f z;
#pragma unroll
  for (int i = 0; i < 8; ++i) z[i] = 0.0f;
  return z;
}

// A-matrix fragment (16x32 bf16, M x K): lane<16 -> row = base_row+lane, K in
// {k0..k0+15}; lane>=16 -> row = base_row+lane-16, K in
// {k0+16..k0+31}. Storage row-major, rowStride elements.
__device__ __forceinline__ v16bf load_afrag(const __bf16* base, int row, int rowStride,
                                            int k0, int lane) {
  const __bf16* p = base + (size_t)(row + (lane & 15)) * rowStride + k0 + ((lane >> 4) << 4);
  Frag16 f;
  f.q[0] = *(const uint4*)p;
  f.q[1] = *(const uint4*)(p + 8);
  return f.v;
}

// B-matrix fragment (32x16 bf16, K x N): lane holds column col+lane%16,
// K = k0 + 16*(lane/16) + {0..15}; storage row-major [N][K].
__device__ __forceinline__ v16bf load_bfrag(const __bf16* base, int col, int rowStride,
                                            int k0, int lane) {
  const __bf16* p = base + (size_t)(col + (lane & 15)) * rowStride + k0 + ((lane >> 4) << 4);
  Frag16 f;
  f.q[0] = ((const uint4*)p)[0];
  f.q[1] = ((const uint4*)p)[1];
  return f.v;
}

// gfx1250 async copy global->LDS (tracked by ASYNCcnt), 16B per lane.
__device__ __forceinline__ void async_b128(unsigned int lds_off, unsigned long long gaddr) {
  asm volatile("global_load_async_to_lds_b128 %0, %1, off"
               :: "v"(lds_off), "v"(gaddr) : "memory");
}

__device__ __forceinline__ unsigned int lds_addr_of(const void* p) {
  // Flat LDS addresses carry the wave-relative byte offset in the low 32 bits.
  return (unsigned int)(unsigned long long)(uintptr_t)p;
}

// ---------------------------------------------------------------------------
// Kernel 0: convert the four weight matrices to bf16 (one-time, tiny).
// ---------------------------------------------------------------------------
__global__ __launch_bounds__(256) void cvt_kernel(
    const float* __restrict__ wt, const float* __restrict__ wp,
    const float* __restrict__ wg, const float* __restrict__ wo,
    __bf16* __restrict__ wtb, __bf16* __restrict__ wpb,
    __bf16* __restrict__ wgb, __bf16* __restrict__ wob) {
  const int i = blockIdx.x * 256 + threadIdx.x;
  if (i < CI_N * C_IN) {
    wtb[i] = (__bf16)wt[i];
    wpb[i] = (__bf16)wp[i];
    wgb[i] = (__bf16)wg[i];
    wob[i] = (__bf16)wo[i];
  }
}

// ---------------------------------------------------------------------------
// Kernel 1: WMMA projections. Per block: 16 s-positions, all 128 out-channels,
// all three weights. x tile transposed+converted into LDS; A-fragments shared
// by 4 waves; B = bf16 weights (compile-time weight selection so all fragment
// loads stay global_load_b128, not flat). Each wave: o-tiles {wid*2, wid*2+1}
// for each of theta/phi/g.
//   theta[n][s][o], phiT[n][s][o] (bf16, [S][CI]);  gT[n][o][s] (bf16, [CI][S])
// block: 128; grid: (S/16, NB)
// ---------------------------------------------------------------------------
__global__ __launch_bounds__(128) void proj_kernel(
    const float* __restrict__ x,
    const __bf16* __restrict__ wtb, const __bf16* __restrict__ wpb,
    const __bf16* __restrict__ wgb,
    const float* __restrict__ bt, const float* __restrict__ bp,
    const float* __restrict__ bg,
    __bf16* __restrict__ theta, __bf16* __restrict__ phiT, __bf16* __restrict__ gT) {
  __shared__ __align__(16) __bf16 xs[16][C_IN];   // x^T tile, 8KB
  const int n = blockIdx.y;
  const int sbase = blockIdx.x * 16;
  const int tid = threadIdx.x;
  const int lane = tid & 31;
  const int wid = tid >> 5;
  const int l16 = lane & 15;
  const int lhi = lane >> 4;
  const float* xn = x + (size_t)n * C_IN * S_TOT;

  // load + transpose + convert: 4096 elements, 32 per thread
#pragma unroll
  for (int i = 0; i < 32; ++i) {
    const int idx = tid + i * 128;
    const int c = idx >> 4, s = idx & 15;
    xs[s][c] = (__bf16)xn[(size_t)c * S_TOT + sbase + s];
  }
  __syncthreads();

  // resident A fragments (x^T), K = 256 -> 8 chunks of 32
  v16bf ax[8];
#pragma unroll
  for (int kc = 0; kc < 8; ++kc)
    ax[kc] = load_afrag(&xs[0][0], 0, C_IN, kc * 32, lane);

#pragma unroll
  for (int t = 0; t < 2; ++t) {
    const int ot = wid * 2 + t;        // o-tile 0..7
    const int o = ot * 16 + l16;       // D column

    // ---- theta ----
    {
      v8f d = zero8();
#pragma unroll
      for (int kc = 0; kc < 8; ++kc)
        d = wmma_bf16(ax[kc], load_bfrag(wtb, ot * 16, C_IN, kc * 32, lane), d);
      const float bval = bt[o];
#pragma unroll
      for (int e = 0; e < 8; ++e) {
        const int s = sbase + e + (lhi ? 8 : 0);
        theta[((size_t)n * S_TOT + s) * CI_N + o] = (__bf16)(d[e] + bval);
      }
    }
    // ---- phi ----
    {
      v8f d = zero8();
#pragma unroll
      for (int kc = 0; kc < 8; ++kc)
        d = wmma_bf16(ax[kc], load_bfrag(wpb, ot * 16, C_IN, kc * 32, lane), d);
      const float bval = bp[o];
#pragma unroll
      for (int e = 0; e < 8; ++e) {
        const int s = sbase + e + (lhi ? 8 : 0);
        phiT[((size_t)n * S_TOT + s) * CI_N + o] = (__bf16)(d[e] + bval);
      }
    }
    // ---- g (transposed store) ----
    {
      v8f d = zero8();
#pragma unroll
      for (int kc = 0; kc < 8; ++kc)
        d = wmma_bf16(ax[kc], load_bfrag(wgb, ot * 16, C_IN, kc * 32, lane), d);
      const float bval = bg[o];
#pragma unroll
      for (int e = 0; e < 8; ++e) {
        const int s = sbase + e + (lhi ? 8 : 0);
        gT[((size_t)n * CI_N + o) * S_TOT + s] = (__bf16)(d[e] + bval);
      }
    }
  }
}

// ---------------------------------------------------------------------------
// Kernel 2: flash-attention core. Key tiles (32 keys: phi^T 8KB + g^T 8KB)
// staged into double-buffered LDS with global_load_async_to_lds_b128
// (8 instructions/wave/tile, ASYNCcnt), WMMA fragments come from LDS.
// Each wave owns 16 queries; S^T orientation keeps softmax lane-local + one
// xor-16 shuffle. Output y[n][q][c] bf16 (B-fragment-ready for kernel 3).
// block: 128 (4 waves); grid: (S/64, NB)
// ---------------------------------------------------------------------------
__global__ __launch_bounds__(128) void flash_kernel(
    const __bf16* __restrict__ theta, const __bf16* __restrict__ phiT,
    const __bf16* __restrict__ gT, __bf16* __restrict__ yb) {
  __shared__ __align__(16) __bf16 phis[2][32][CI_N];   // 2 x 8KB
  __shared__ __align__(16) __bf16 gs[2][CI_N][32];     // 2 x 8KB

  const int tid   = threadIdx.x;
  const int lane  = tid & 31;
  const int wid   = tid >> 5;
  const int n     = blockIdx.y;
  const int qbase = blockIdx.x * 64 + wid * 16;
  const int l16   = lane & 15;
  const int lhi   = lane >> 4;

  const __bf16* thetaN = theta + (size_t)n * S_TOT * CI_N;
  const __bf16* phiN   = phiT  + (size_t)n * S_TOT * CI_N;
  const __bf16* gN     = gT    + (size_t)n * CI_N  * S_TOT;

  const unsigned int phis_base = lds_addr_of(&phis[0][0][0]);
  const unsigned int gs_base   = lds_addr_of(&gs[0][0][0]);

  // issue one key tile's async copies (8 x b128 instructions per wave)
  auto issue_tile = [&](int sb, int buf) {
#pragma unroll
    for (int i = 0; i < 4; ++i) {                       // phi^T tile 32x128
      const int chunk = i * 128 + tid;                  // 0..511
      const int srow = chunk >> 4, cc = (chunk & 15) << 3;
      async_b128(phis_base + (unsigned int)(buf * 32 * CI_N + srow * CI_N + cc) * 2u,
                 (unsigned long long)(uintptr_t)(phiN + (size_t)(sb + srow) * CI_N + cc));
    }
#pragma unroll
    for (int i = 0; i < 4; ++i) {                       // g^T tile 128x32
      const int chunk = i * 128 + tid;
      const int crow = chunk >> 2, ss = (chunk & 3) << 3;
      async_b128(gs_base + (unsigned int)(buf * CI_N * 32 + crow * 32 + ss) * 2u,
                 (unsigned long long)(uintptr_t)(gN + (size_t)crow * S_TOT + sb + ss));
    }
  };

  // resident query B-fragments
  v16bf bq[4];
#pragma unroll
  for (int kc = 0; kc < 4; ++kc)
    bq[kc] = load_bfrag(thetaN, qbase, CI_N, kc * 32, lane);

  v8f O[8];
#pragma unroll
  for (int t = 0; t < 8; ++t) O[t] = zero8();
  float m_run = -INFINITY;
  float l_run = 0.0f;

  const int J = S_TOT / 32;
  issue_tile(0, 0);

  for (int j = 0; j < J; ++j) {
    const int buf = j & 1;
    if (j + 1 < J) {
      issue_tile((j + 1) * 32, 1 - buf);
      asm volatile("s_wait_asynccnt 0x8" ::: "memory");  // tile j landed
    } else {
      asm volatile("s_wait_asynccnt 0x0" ::: "memory");
    }
    __syncthreads();                                     // tile j visible to all

    const __bf16* pt = &phis[buf][0][0];
    const __bf16* gt = &gs[buf][0][0];

    // S^T tiles (keys x 16 queries)
    v8f s0 = zero8(), s1 = zero8();
#pragma unroll
    for (int kc = 0; kc < 4; ++kc) {
      s0 = wmma_bf16(load_afrag(pt, 0,  CI_N, kc * 32, lane), bq[kc], s0);
      s1 = wmma_bf16(load_afrag(pt, 16, CI_N, kc * 32, lane), bq[kc], s1);
    }

    // online softmax (lane owns one query column; partner lane = xor 16)
    float p0[8], p1[8];
    float mt = -INFINITY;
#pragma unroll
    for (int e = 0; e < 8; ++e) {
      p0[e] = s0[e] * ATT_SCALE;
      p1[e] = s1[e] * ATT_SCALE;
      mt = fmaxf(mt, fmaxf(p0[e], p1[e]));
    }
    mt = fmaxf(mt, __shfl_xor(mt, 16, 32));
    const float mn = fmaxf(m_run, mt);
    const float alpha = __expf(m_run - mn);
    m_run = mn;

    float rs = 0.0f;
#pragma unroll
    for (int e = 0; e < 8; ++e) {
      p0[e] = __expf(p0[e] - mn);
      p1[e] = __expf(p1[e] - mn);
      rs += p0[e] + p1[e];
    }
    l_run = l_run * alpha + rs;

    // re-pack P^T (32 keys x 16 queries) into a B-fragment
    Frag16 pb;
#pragma unroll
    for (int e = 0; e < 8; ++e) {
      const float x0 = __shfl_xor(p0[e], 16, 32);
      const float x1 = __shfl_xor(p1[e], 16, 32);
      pb.e[e]     = (__bf16)(lhi ? x1 : p0[e]);
      pb.e[e + 8] = (__bf16)(lhi ? p1[e] : x0);
    }

    // y^T accumulation, A = g tile rows (c), K = 32 keys
#pragma unroll
    for (int ct = 0; ct < 8; ++ct) {
      v16bf ag = load_afrag(gt, ct * 16, 32, 0, lane);
#pragma unroll
      for (int e = 0; e < 8; ++e) O[ct][e] *= alpha;
      O[ct] = wmma_bf16(ag, pb.v, O[ct]);
    }
    __syncthreads();   // everyone done with buf before it is overwritten
  }

  const float l_tot = l_run + __shfl_xor(l_run, 16, 32);
  const float inv = 1.0f / l_tot;

  // store y[n][q][c] bf16 (row-major [S][CI])
#pragma unroll
  for (int ct = 0; ct < 8; ++ct) {
#pragma unroll
    for (int e = 0; e < 8; ++e) {
      const int c = ct * 16 + e + (lhi ? 8 : 0);
      yb[((size_t)n * S_TOT + qbase + l16) * CI_N + c] = (__bf16)(O[ct][e] * inv);
    }
  }
}

// ---------------------------------------------------------------------------
// Kernel 3: WMMA output projection + bias + residual.
//   out[n][o][s] = x[n][o][s] + b_out[o] + sum_c w_out[o][c] * y[n][s][c]
// A = w_out bf16 [o][c], B = y bf16 [s][c] column fragments. K = 128.
// block: 128 (4 waves, 4 o-tiles each); grid: (S/16, NB)
// ---------------------------------------------------------------------------
__global__ __launch_bounds__(128) void outproj_kernel(
    const float* __restrict__ x, const __bf16* __restrict__ wob,
    const float* __restrict__ bout, const __bf16* __restrict__ yb,
    float* __restrict__ out) {
  const int tid = threadIdx.x;
  const int lane = tid & 31;
  const int wid = tid >> 5;
  const int n = blockIdx.y;
  const int sbase = blockIdx.x * 16;
  const int l16 = lane & 15;
  const int lhi = lane >> 4;

  const __bf16* ybN = yb + (size_t)n * S_TOT * CI_N;

  v16bf by[4];
#pragma unroll
  for (int kc = 0; kc < 4; ++kc)
    by[kc] = load_bfrag(ybN, sbase, CI_N, kc * 32, lane);

#pragma unroll
  for (int k = 0; k < 4; ++k) {
    const int ot = wid * 4 + k;       // 0..15 (o-tile of 16)
    v8f d = zero8();
#pragma unroll
    for (int kc = 0; kc < 4; ++kc)
      d = wmma_bf16(load_afrag(wob, ot * 16, CI_N, kc * 32, lane), by[kc], d);

#pragma unroll
    for (int e = 0; e < 8; ++e) {
      const int o = ot * 16 + e + (lhi ? 8 : 0);
      const size_t idx = ((size_t)n * C_IN + o) * S_TOT + sbase + l16;
      out[idx] = x[idx] + bout[o] + d[e];
    }
  }
}

// ---------------------------------------------------------------------------
extern "C" void kernel_launch(void* const* d_in, const int* in_sizes, int n_in,
                              void* d_out, int out_size, void* d_ws, size_t ws_size,
                              hipStream_t stream) {
  (void)in_sizes; (void)n_in; (void)out_size; (void)ws_size;
  const float* x    = (const float*)d_in[0];
  const float* w_g  = (const float*)d_in[1];
  const float* b_g  = (const float*)d_in[2];
  const float* w_t  = (const float*)d_in[3];
  const float* b_t  = (const float*)d_in[4];
  const float* w_p  = (const float*)d_in[5];
  const float* b_p  = (const float*)d_in[6];
  const float* w_o  = (const float*)d_in[7];
  const float* b_o  = (const float*)d_in[8];
  float* out = (float*)d_out;

  const size_t projBytes = (size_t)NB * S_TOT * CI_N * sizeof(__bf16); // 4 MiB
  const size_t wBytes = (size_t)CI_N * C_IN * sizeof(__bf16);          // 64 KiB
  char* ws = (char*)d_ws;
  __bf16* theta = (__bf16*)(ws);
  __bf16* phiT  = (__bf16*)(ws + projBytes);
  __bf16* gT    = (__bf16*)(ws + 2 * projBytes);
  __bf16* yb    = (__bf16*)(ws + 3 * projBytes);
  __bf16* wtb   = (__bf16*)(ws + 4 * projBytes);
  __bf16* wpb   = (__bf16*)(ws + 4 * projBytes + wBytes);
  __bf16* wgb   = (__bf16*)(ws + 4 * projBytes + 2 * wBytes);
  __bf16* wob   = (__bf16*)(ws + 4 * projBytes + 3 * wBytes);

  cvt_kernel<<<dim3((CI_N * C_IN + 255) / 256), 256, 0, stream>>>(
      w_t, w_p, w_g, w_o, wtb, wpb, wgb, wob);
  proj_kernel<<<dim3(S_TOT / 16, NB), 128, 0, stream>>>(
      x, wtb, wpb, wgb, b_t, b_p, b_g, theta, phiT, gT);
  flash_kernel<<<dim3(S_TOT / 64, NB), 128, 0, stream>>>(theta, phiT, gT, yb);
  outproj_kernel<<<dim3(S_TOT / 16, NB), 128, 0, stream>>>(x, wob, b_o, yb, out);
}
